// LSTMModel_56083682951409
// MI455X (gfx1250) — compile-verified, matrix operand order checked by reading, not run on hardware
//
#include <hip/hip_runtime.h>
#include <hip/hip_bf16.h>

// ---------------------------------------------------------------------------
// LSTM LM on gfx1250: bf16 WMMA (v_wmma_f32_16x16x32_bf16) for all matmuls,
// TDM (tensor_load_to_lds + s_wait_tensorcnt) to prefetch the per-step xg
// tile into LDS during the sequential scan.
//  B=128, T=1024, E=H=256, 4H=1024, VOCAB=128, LAYERS=4, T-chunk=128.
// ---------------------------------------------------------------------------

typedef __attribute__((ext_vector_type(16))) __bf16 v16bf;
typedef __attribute__((ext_vector_type(8)))  __bf16 v8bf;
typedef __attribute__((ext_vector_type(8)))  float  v8f;
typedef unsigned int u32x4 __attribute__((ext_vector_type(4)));
typedef int          i32x4 __attribute__((ext_vector_type(4)));
typedef int          i32x8 __attribute__((ext_vector_type(8)));

#define HID   256
#define G4H   1024
#define BATCH 128
#define TSEQ  1024
#define TCH   128   // timesteps per chunk
#define NCHUNK (TSEQ / TCH)

// Load a 16x32 bf16 WMMA fragment row/col for this lane.
// prow: pointer to this lane's row (A) / column-as-row (B), contiguous in K.
// kpos = kb*32 + khalf  (khalf = (lane>=16)?8:0). Two 16B chunks: K and K+16.
__device__ __forceinline__ v16bf frag_ld(const __bf16* prow, int kpos) {
    v8bf lo = *reinterpret_cast<const v8bf*>(prow + kpos);
    v8bf hi = *reinterpret_cast<const v8bf*>(prow + kpos + 16);
    v16bf r;
#pragma unroll
    for (int j = 0; j < 8; ++j) { r[j] = lo[j]; r[j + 8] = hi[j]; }
    return r;
}

__device__ __forceinline__ float sigm_f(float x) {
    return 1.0f / (1.0f + __expf(-x));
}
__device__ __forceinline__ float tanh_f(float x) {
    float e = __expf(-2.0f * x);
    return (1.0f - e) / (1.0f + e);
}

// ---------------------------------------------------------------------------
// TDM: DMA a 2D tile [16 rows x 1024 f32], row stride 131072 f32, from global
// xg into LDS at lds_addr. D# packed per CDNA5 ISA §8.3/8.4 (type=2, count=1,
// data_size=4B). Tracked by TENSORcnt.
// ---------------------------------------------------------------------------
__device__ __forceinline__ void tdm_load_xg_tile(unsigned lds_addr,
                                                 const float* gsrc) {
    unsigned long long ga = (unsigned long long)(uintptr_t)gsrc;
    u32x4 g0;
    g0[0] = 1u;                                              // count=1 (valid)
    g0[1] = lds_addr;                                        // LDS byte addr
    g0[2] = (unsigned)(ga & 0xFFFFFFFFu);                    // global_addr lo
    g0[3] = (unsigned)((ga >> 32) & 0x01FFFFFFu) | 0x80000000u; // hi | type=2
    i32x8 g1;
    g1[0] = 0x00020000;          // wg_mask=0, data_size=2 (4B), no flags
    g1[1] = (int)(1024u << 16);  // atomic_bar=0 | tensor_dim0 lo16 = 1024
    g1[2] = (int)(16u << 16);    // tensor_dim0 hi=0 | tensor_dim1 lo16 = 16
    g1[3] = (int)(1024u << 16);  // tensor_dim1 hi=0 | tile_dim0 = 1024
    g1[4] = 16;                  // tile_dim1 = 16, tile_dim2 = 0
    g1[5] = 131072;              // tensor_dim0_stride lo32 (elements)
    g1[6] = 0;                   // stride hi | tensor_dim1_stride lo
    g1[7] = 0;
    i32x4 gz = {0, 0, 0, 0};
#if defined(__clang_major__) && (__clang_major__ >= 23)
    i32x8 gz8 = {0, 0, 0, 0, 0, 0, 0, 0};
    __builtin_amdgcn_tensor_load_to_lds(g0, g1, gz, gz, gz8, 0);
#else
    __builtin_amdgcn_tensor_load_to_lds(g0, g1, gz, gz, 0);
#endif
}

// ---------------------------------------------------------------------------
// Prep: fp32 -> bf16 weight conversion
// ---------------------------------------------------------------------------
__global__ void cvt_bf16_k(const float* __restrict__ wih,
                           const float* __restrict__ whh,
                           const float* __restrict__ fcw,
                           __bf16* __restrict__ wihb,
                           __bf16* __restrict__ whhb,
                           __bf16* __restrict__ fcwb) {
    int i = blockIdx.x * 256 + threadIdx.x;           // grid covers 4*1024*256
    wihb[i] = (__bf16)wih[i];
    whhb[i] = (__bf16)whh[i];
    if (i < 128 * 256) fcwb[i] = (__bf16)fcw[i];
}

// Embedding gather -> bf16 activations, row = b*T + t (b-major)
__global__ void embed_k(const int* __restrict__ x,
                        const float* __restrict__ tab,
                        __bf16* __restrict__ act) {
    int r = blockIdx.x;                                // 0..B*T-1
    int e = threadIdx.x;                               // 0..255
    int xi = x[r];
    act[(size_t)r * HID + e] = (__bf16)tab[xi * HID + e];
}

// ---------------------------------------------------------------------------
// xg = act_chunk @ Wih^T + (b_ih + b_hh)   for one T-chunk of one layer.
// Output rows r = b*TCH + tl  (16384 rows), cols 0..1023.  K = 256.
// grid 8192 x 256 threads (8 waves/block), one 16x16 tile per wave.
// ---------------------------------------------------------------------------
__global__ void xg_gemm_k(const __bf16* __restrict__ act,   // [B*T, 256]
                          const __bf16* __restrict__ wih,   // [1024, 256]
                          const float* __restrict__ bih,    // [1024]
                          const float* __restrict__ bhh,    // [1024]
                          float* __restrict__ xg,           // [B*TCH, 1024]
                          int chunk) {
    int wave = threadIdx.x >> 5;
    int lane = threadIdx.x & 31;
    int tile = blockIdx.x * 8 + wave;                  // 1024*64 tiles
    int mt = tile >> 6;
    int nt = tile & 63;

    int ml    = lane & 15;
    int khalf = (lane >> 4) * 8;

    int r  = mt * 16 + ml;                             // chunk-local row
    int b  = r >> 7;                                   // /TCH
    int tl = r & 127;
    const __bf16* arow = act + ((size_t)b * TSEQ + (size_t)chunk * TCH + tl) * HID;
    const __bf16* brow = wih + (size_t)(nt * 16 + ml) * HID;

    v8f acc = {};
#pragma unroll
    for (int kb = 0; kb < 8; ++kb) {
        v16bf a  = frag_ld(arow, kb * 32 + khalf);
        v16bf bm = frag_ld(brow, kb * 32 + khalf);
        acc = __builtin_amdgcn_wmma_f32_16x16x32_bf16(false, a, false, bm,
                                                      (short)0, acc, false, false);
    }

    int col   = nt * 16 + ml;
    float bias = bih[col] + bhh[col];
    int rbase = mt * 16 + (lane >> 4) * 8;
#pragma unroll
    for (int i = 0; i < 8; ++i)
        xg[(size_t)(rbase + i) * G4H + col] = acc[i] + bias;
}

// ---------------------------------------------------------------------------
// Sequential LSTM scan over one T-chunk. 8 blocks (16 batch rows each),
// 512 threads = 16 waves; wave w owns gate columns [64w, 64w+64).
// Per step: 16x1024 = h @ Whh^T via 4 tiles x 8 K-blocks of WMMA; the xg
// tile for step t+1 is TDM-prefetched into LDS (double buffered) while the
// WMMAs for step t run; gates combine cross-wave through LDS.
// ---------------------------------------------------------------------------
__global__ void lstm_scan_k(const float* __restrict__ xg,     // [B*TCH, 1024]
                            const __bf16* __restrict__ whh,   // [1024, 256]
                            const float* __restrict__ h0,     // [B, 256] slice
                            const float* __restrict__ c0,
                            float* __restrict__ hstate,       // [B, 256] ws
                            float* __restrict__ cstate,
                            __bf16* __restrict__ act_out,     // [B*T, 256]
                            float* __restrict__ hN,           // [B, 256] slice
                            float* __restrict__ cN,
                            int chunk, int lastchunk) {
    __shared__ __align__(16) float  g_lds[16 * G4H];      //  64 KB
    __shared__ __align__(16) float  xg_lds[2][16 * G4H];  // 128 KB (TDM dest)
    __shared__ __align__(16) __bf16 hbf[16 * HID];        //   8 KB
    __shared__ __align__(16) float  hf[16 * HID];         //  16 KB
    __shared__ __align__(16) float  cf[16 * HID];         //  16 KB

    const int b0  = blockIdx.x * 16;
    const int tid = threadIdx.x;

    // Load carried state (h0/c0 on the first chunk).
    for (int p = tid; p < 16 * HID; p += 512) {
        int b = p >> 8, hh = p & 255;
        float hv, cv;
        if (chunk == 0) { hv = h0[(b0 + b) * HID + hh]; cv = c0[(b0 + b) * HID + hh]; }
        else            { hv = hstate[(b0 + b) * HID + hh]; cv = cstate[(b0 + b) * HID + hh]; }
        hf[p] = hv; cf[p] = cv; hbf[p] = (__bf16)hv;
    }

    const int wave  = tid >> 5;
    const int lane  = tid & 31;
    const int ml    = lane & 15;
    const int khalf = (lane >> 4) * 8;
    const int cbase = wave * 64;
    const __bf16* aprow = hbf + ml * HID;
    const __bf16* brow0 = whh + (size_t)(cbase +  0 + ml) * HID;
    const __bf16* brow1 = whh + (size_t)(cbase + 16 + ml) * HID;
    const __bf16* brow2 = whh + (size_t)(cbase + 32 + ml) * HID;
    const __bf16* brow3 = whh + (size_t)(cbase + 48 + ml) * HID;
    const int rb = (lane >> 4) * 8;

    // Kick off the TDM prefetch of the xg tile for step 0 into buffer 0.
    const float* xg_row0 = xg + (size_t)b0 * TCH * G4H;   // row (b0*TCH + tl)
    if (wave == 0) {
        tdm_load_xg_tile((unsigned)(uintptr_t)(void*)&xg_lds[0][0], xg_row0);
    }
    __syncthreads();

    for (int tl = 0; tl < TCH; ++tl) {
        const int cur = tl & 1;
        // Issue TDM for step tl+1 (clamped on the last step so the in-order
        // TENSORcnt wait below still covers step tl's transfer).
        if (wave == 0) {
            int tln = (tl + 1 < TCH) ? tl + 1 : TCH - 1;
            tdm_load_xg_tile((unsigned)(uintptr_t)(void*)&xg_lds[cur ^ 1][0],
                             xg_row0 + (size_t)tln * G4H);
        }

        v8f a0 = {}, a1 = {}, a2 = {}, a3 = {};
#pragma unroll
        for (int kb = 0; kb < 8; ++kb) {
            int kpos = kb * 32 + khalf;
            v16bf a = frag_ld(aprow, kpos);
            v16bf w0 = frag_ld(brow0, kpos);
            a0 = __builtin_amdgcn_wmma_f32_16x16x32_bf16(false, a, false, w0,
                                                         (short)0, a0, false, false);
            v16bf w1 = frag_ld(brow1, kpos);
            a1 = __builtin_amdgcn_wmma_f32_16x16x32_bf16(false, a, false, w1,
                                                         (short)0, a1, false, false);
            v16bf w2 = frag_ld(brow2, kpos);
            a2 = __builtin_amdgcn_wmma_f32_16x16x32_bf16(false, a, false, w2,
                                                         (short)0, a2, false, false);
            v16bf w3 = frag_ld(brow3, kpos);
            a3 = __builtin_amdgcn_wmma_f32_16x16x32_bf16(false, a, false, w3,
                                                         (short)0, a3, false, false);
        }

        // Ensure this step's xg tile (issued last iteration / preloop) landed:
        // 2 TDM ops outstanding, in-order -> waiting to <=1 retires step tl's.
        if (wave == 0) __builtin_amdgcn_s_wait_tensorcnt(1);
        __syncthreads();

        // epilogue: add xg (from LDS) and stage pre-activations in LDS
#pragma unroll
        for (int i = 0; i < 8; ++i) {
            int bl = rb + i;
            int c0c = cbase +  0 + ml;
            int c1c = cbase + 16 + ml;
            int c2c = cbase + 32 + ml;
            int c3c = cbase + 48 + ml;
            g_lds[bl * G4H + c0c] = a0[i] + xg_lds[cur][bl * G4H + c0c];
            g_lds[bl * G4H + c1c] = a1[i] + xg_lds[cur][bl * G4H + c1c];
            g_lds[bl * G4H + c2c] = a2[i] + xg_lds[cur][bl * G4H + c2c];
            g_lds[bl * G4H + c3c] = a3[i] + xg_lds[cur][bl * G4H + c3c];
        }
        __syncthreads();

        // gates: 4096 (b,hid) pairs / 512 threads
        int t = chunk * TCH + tl;
#pragma unroll
        for (int j = 0; j < 8; ++j) {
            int p  = tid + j * 512;
            int b  = p >> 8;
            int hh = p & 255;
            float gi = g_lds[b * G4H + hh];
            float gf = g_lds[b * G4H + 256 + hh];
            float gg = g_lds[b * G4H + 512 + hh];
            float go = g_lds[b * G4H + 768 + hh];
            float ii = sigm_f(gi);
            float ff = sigm_f(gf);
            float cc = tanh_f(gg);
            float oo = sigm_f(go);
            float cv = ff * cf[p] + ii * cc;
            float hv = oo * tanh_f(cv);
            cf[p] = cv;
            hf[p] = hv;
            hbf[p] = (__bf16)hv;
            act_out[((size_t)(b0 + b) * TSEQ + t) * HID + hh] = (__bf16)hv;
        }
        __syncthreads();
    }

    if (wave == 0) __builtin_amdgcn_s_wait_tensorcnt(0);  // drain the tail TDM

    // persist carried state (+ final h/c outputs on the last chunk)
    for (int p = tid; p < 16 * HID; p += 512) {
        int b = p >> 8, hh = p & 255;
        hstate[(b0 + b) * HID + hh] = hf[p];
        cstate[(b0 + b) * HID + hh] = cf[p];
        if (lastchunk) {
            hN[(b0 + b) * HID + hh] = hf[p];
            cN[(b0 + b) * HID + hh] = cf[p];
        }
    }
}

// ---------------------------------------------------------------------------
// Final FC: out[B*T,128] = hs @ fc_w^T + fc_b.  8192 M-tiles x 8 N-tiles.
// ---------------------------------------------------------------------------
__global__ void fc_gemm_k(const __bf16* __restrict__ act,   // [B*T, 256]
                          const __bf16* __restrict__ fcw,   // [128, 256]
                          const float* __restrict__ fcb,    // [128]
                          float* __restrict__ out) {        // [B*T, 128]
    int wave = threadIdx.x >> 5;
    int lane = threadIdx.x & 31;
    int tile = blockIdx.x * 8 + wave;                  // 8192*8 tiles
    int mt = tile >> 3;
    int nt = tile & 7;

    int ml    = lane & 15;
    int khalf = (lane >> 4) * 8;
    const __bf16* arow = act + (size_t)(mt * 16 + ml) * HID;
    const __bf16* brow = fcw + (size_t)(nt * 16 + ml) * HID;

    v8f acc = {};
#pragma unroll
    for (int kb = 0; kb < 8; ++kb) {
        v16bf a  = frag_ld(arow, kb * 32 + khalf);
        v16bf bm = frag_ld(brow, kb * 32 + khalf);
        acc = __builtin_amdgcn_wmma_f32_16x16x32_bf16(false, a, false, bm,
                                                      (short)0, acc, false, false);
    }
    int col  = nt * 16 + ml;
    float bias = fcb[col];
    int row0 = mt * 16 + (lane >> 4) * 8;
#pragma unroll
    for (int i = 0; i < 8; ++i)
        out[(size_t)(row0 + i) * 128 + col] = acc[i] + bias;
}

// ---------------------------------------------------------------------------
extern "C" void kernel_launch(void* const* d_in, const int* in_sizes, int n_in,
                              void* d_out, int out_size, void* d_ws, size_t ws_size,
                              hipStream_t stream) {
    (void)in_sizes; (void)n_in; (void)out_size; (void)ws_size;

    const int*   x   = (const int*)  d_in[0];
    const float* h0  = (const float*)d_in[1];
    const float* c0  = (const float*)d_in[2];
    const float* tab = (const float*)d_in[3];
    const float* Wih = (const float*)d_in[4];
    const float* Whh = (const float*)d_in[5];
    const float* bih = (const float*)d_in[6];
    const float* bhh = (const float*)d_in[7];
    const float* fcw = (const float*)d_in[8];
    const float* fcb = (const float*)d_in[9];

    float* out = (float*)d_out;
    float* hN  = out + (size_t)BATCH * TSEQ * 128;     // 16,777,216
    float* cN  = hN + (size_t)4 * BATCH * HID;         // +131,072

    char* p = (char*)d_ws;
    auto carve = [&](size_t bytes) {
        void* r = (void*)p;
        p += (bytes + 255) & ~(size_t)255;
        return r;
    };
    __bf16* act0   = (__bf16*)carve((size_t)BATCH * TSEQ * HID * 2);
    __bf16* act1   = (__bf16*)carve((size_t)BATCH * TSEQ * HID * 2);
    float*  xg     = (float*) carve((size_t)BATCH * TCH * G4H * 4);
    __bf16* wihb   = (__bf16*)carve((size_t)4 * G4H * HID * 2);
    __bf16* whhb   = (__bf16*)carve((size_t)4 * G4H * HID * 2);
    __bf16* fcwb   = (__bf16*)carve((size_t)128 * HID * 2);
    float*  hstate = (float*) carve((size_t)BATCH * HID * 4);
    float*  cstate = (float*) carve((size_t)BATCH * HID * 4);

    cvt_bf16_k<<<4 * G4H * HID / 256, 256, 0, stream>>>(Wih, Whh, fcw, wihb, whhb, fcwb);
    embed_k<<<BATCH * TSEQ, HID, 0, stream>>>(x, tab, act0);

    const __bf16* ain = act0;
    __bf16*       aout = act1;
    for (int l = 0; l < 4; ++l) {
        for (int ch = 0; ch < NCHUNK; ++ch) {
            xg_gemm_k<<<8192, 256, 0, stream>>>(
                ain, wihb + (size_t)l * G4H * HID,
                bih + l * G4H, bhh + l * G4H, xg, ch);
            lstm_scan_k<<<8, 512, 0, stream>>>(
                xg, whhb + (size_t)l * G4H * HID,
                h0 + (size_t)l * BATCH * HID, c0 + (size_t)l * BATCH * HID,
                hstate, cstate, aout,
                hN + (size_t)l * BATCH * HID, cN + (size_t)l * BATCH * HID,
                ch, (ch == NCHUNK - 1) ? 1 : 0);
        }
        const __bf16* t2 = ain; ain = aout; aout = (__bf16*)t2;
    }

    fc_gemm_k<<<8192, 256, 0, stream>>>(ain, fcwb, fcb, out);
}
